// Encoder_10325101380015
// MI455X (gfx1250) — matrix operand order repaired and checked by reading, not compiled, not run on hardware
//
#include <hip/hip_runtime.h>
#include <stdint.h>

// Problem sizes (fixed by the reference)
#define NN   100000       // nodes
#define NE   250000       // edges
#define NB   2000         // graphs
#define FIN  75
#define HID  128
#define MT_E (NE / 16)    // 15625 edge tiles of 16
#define MT_N (NN / 16)    // 6250 node tiles of 16

typedef __attribute__((ext_vector_type(8)))  float    v8f;
typedef __attribute__((ext_vector_type(16))) __bf16   v16bf;
typedef __attribute__((ext_vector_type(8)))  __bf16   v8bf;
typedef __attribute__((ext_vector_type(4)))  unsigned v4u;

union ABu { v16bf v; v8bf h[2]; unsigned short u16[16]; unsigned d[8]; };

static __device__ __forceinline__ unsigned short f2bf(float f) {
  unsigned u = __float_as_uint(f);
  u += 0x7FFFu + ((u >> 16) & 1u);          // round-to-nearest-even
  return (unsigned short)(u >> 16);
}
static __device__ __forceinline__ float bf2f(unsigned short h) {
  return __uint_as_float((unsigned)h << 16);
}
static __device__ __forceinline__ float sigm(float x) { return 1.f / (1.f + __expf(-x)); }

// order-preserving float -> uint key (for atomicMax-based segment max)
static __device__ __forceinline__ unsigned fkey(float f) {
  unsigned u = __float_as_uint(f);
  return (u & 0x80000000u) ? ~u : (u | 0x80000000u);
}
static __device__ __forceinline__ float fdec(unsigned k) {
  return __uint_as_float((k & 0x80000000u) ? (k & 0x7FFFFFFFu) : ~k);
}

// Build a bf16 16x32 WMMA A-fragment from an f32 row of 32 (contiguous layout:
// lanes<16: halves 0..15 = K0..15; lanes>=16: K16..31).
static __device__ __forceinline__ v16bf loadA_f32(const float* rowbase, int lh) {
  ABu u;
  const float4* q = (const float4*)(rowbase + lh * 16);
  float4 a0 = q[0], a1 = q[1], a2 = q[2], a3 = q[3];
  u.u16[0] = f2bf(a0.x);  u.u16[1] = f2bf(a0.y);  u.u16[2] = f2bf(a0.z);  u.u16[3] = f2bf(a0.w);
  u.u16[4] = f2bf(a1.x);  u.u16[5] = f2bf(a1.y);  u.u16[6] = f2bf(a1.z);  u.u16[7] = f2bf(a1.w);
  u.u16[8] = f2bf(a2.x);  u.u16[9] = f2bf(a2.y);  u.u16[10] = f2bf(a2.z); u.u16[11] = f2bf(a2.w);
  u.u16[12] = f2bf(a3.x); u.u16[13] = f2bf(a3.y); u.u16[14] = f2bf(a3.z); u.u16[15] = f2bf(a3.w);
  return u.v;
}

// ---------------------------------------------------------------------------
// generic zero
__global__ __launch_bounds__(256) void k_zero(unsigned* p, size_t n) {
  for (size_t i = (size_t)blockIdx.x * 256 + threadIdx.x; i < n; i += (size_t)gridDim.x * 256)
    p[i] = 0u;
}

// ---------------------------------------------------------------------------
// lin0: x0 = relu(nf @ lin0_w^T + b)   [NN,75] @ [75,32]
__global__ __launch_bounds__(256) void k_lin0(const float* __restrict__ nf,
                                              const float* __restrict__ w,
                                              const float* __restrict__ b,
                                              float* __restrict__ x0) {
  __shared__ float ws[32 * 76];
  __shared__ float rows[8][76];
  int tid = threadIdx.x;
  for (int i = tid; i < 32 * FIN; i += 256) { int o = i / FIN, k = i % FIN; ws[o * 76 + k] = w[i]; }
  __syncthreads();
  int o = tid & 31, nl = tid >> 5;
  for (int n0 = blockIdx.x * 8; n0 < NN; n0 += gridDim.x * 8) {
    int n = n0 + nl;
    if (n < NN) { for (int k = o; k < FIN; k += 32) rows[nl][k] = nf[(size_t)n * FIN + k]; }
    __syncthreads();
    if (n < NN) {
      float acc = b[o];
      #pragma unroll 15
      for (int k = 0; k < FIN; ++k) acc += rows[nl][k] * ws[o * 76 + k];
      x0[(size_t)n * 32 + o] = fmaxf(acc, 0.f);
    }
    __syncthreads();
  }
}

// ---------------------------------------------------------------------------
// hw = relu(ef @ nn_w1^T + b1)  -> bf16 [NE,128]
__global__ __launch_bounds__(256) void k_hw(const float* __restrict__ ef,
                                            const float* __restrict__ w1,
                                            const float* __restrict__ b1,
                                            unsigned short* __restrict__ hw) {
  size_t total = (size_t)NE * HID;
  for (size_t g = (size_t)blockIdx.x * 256 + threadIdx.x; g < total; g += (size_t)gridDim.x * 256) {
    size_t e = g >> 7; int k = (int)(g & 127);
    const float* er = ef + e * 11;
    const float* wr = w1 + k * 11;
    float acc = b1[k];
    #pragma unroll
    for (int j = 0; j < 11; ++j) acc += er[j] * wr[j];
    hw[g] = f2bf(fmaxf(acc, 0.f));
  }
}

// ---------------------------------------------------------------------------
// Flagship GEMM: edge_w[e, c] = sum_k hw[e,k]*W2[c,k] + b2[c]   (bf16 WMMA)
// Output stored in C-fragment-native order: wpack[tile][ntGlobal][lane][8 bf16]
__global__ __launch_bounds__(256) void k_edge_gemm(const unsigned short* __restrict__ hw,
                                                   const float* __restrict__ w2,
                                                   const float* __restrict__ b2,
                                                   unsigned short* __restrict__ wpack) {
  __shared__ unsigned short w2s[128 * 136];   // slab of 128 cols, padded stride (16B aligned)
  const int slab = blockIdx.y;                // 8 slabs of 128 columns
  const int tid = threadIdx.x;
  for (int idx = tid; idx < 128 * 128; idx += 256) {
    int c = idx >> 7, k = idx & 127;
    w2s[c * 136 + k] = f2bf(w2[((size_t)slab * 128 + c) * 128 + k]);
  }
  __syncthreads();
  const int lane = tid & 31, wave = tid >> 5;
  const int n16 = lane & 15, lh = lane >> 4;
  float bias[8];
  #pragma unroll
  for (int nt = 0; nt < 8; ++nt) bias[nt] = b2[slab * 128 + nt * 16 + n16];

  for (int tile = blockIdx.x * 8 + wave; tile < MT_E; tile += gridDim.x * 8) {
    v16bf a[4];
    const unsigned short* arow = hw + (size_t)(tile * 16 + n16) * HID;
    #pragma unroll
    for (int kc = 0; kc < 4; ++kc) {
      ABu u;
      u.h[0] = *(const v8bf*)(arow + kc * 32 + lh * 16);
      u.h[1] = *(const v8bf*)(arow + kc * 32 + lh * 16 + 8);
      a[kc] = u.v;
    }
    #pragma unroll
    for (int nt = 0; nt < 8; ++nt) {
      v8f c;
      #pragma unroll
      for (int r = 0; r < 8; ++r) c[r] = bias[nt];
      const unsigned short* brow = w2s + (nt * 16 + n16) * 136 + lh * 16;
      #pragma unroll
      for (int kc = 0; kc < 4; ++kc) {
        ABu ub;
        ub.h[0] = *(const v8bf*)(brow + kc * 32);
        ub.h[1] = *(const v8bf*)(brow + kc * 32 + 8);
        c = __builtin_amdgcn_wmma_f32_16x16x32_bf16(false, a[kc], false, ub.v,
                                                    (short)0, c, false, false);
      }
      v4u o4;
      #pragma unroll
      for (int j = 0; j < 4; ++j)
        o4[j] = (unsigned)f2bf(c[2 * j]) | ((unsigned)f2bf(c[2 * j + 1]) << 16);
      size_t idx = ((size_t)tile * 64 + (slab * 8 + nt)) * 32 + lane;
      __builtin_nontemporal_store(o4, (v4u*)wpack + idx);   // coalesced 512B/wave, NT
    }
  }
}

// ---------------------------------------------------------------------------
// degree counts
__global__ __launch_bounds__(256) void k_degree(const int* __restrict__ dst,
                                                float* __restrict__ cnt) {
  for (int g = blockIdx.x * 256 + threadIdx.x; g < NE; g += gridDim.x * 256)
    atomicAdd(&cnt[dst[g]], 1.0f);
}

// ---------------------------------------------------------------------------
// conv: per 16-edge tile (one wave), msg = x[src] * W_e, atomic scatter to msum
__global__ __launch_bounds__(256) void k_conv(const unsigned short* __restrict__ wpack,
                                              const float* __restrict__ x,
                                              const int* __restrict__ esrc,
                                              const int* __restrict__ edst,
                                              float* __restrict__ msum) {
  __shared__ float xs[8][16 * 32];
  __shared__ int   dsts[8][16];
  int tid = threadIdx.x, wave = tid >> 5, lane = tid & 31;
  int n16 = lane & 15, lh = lane >> 4;
  for (int tile = blockIdx.x * 8 + wave; tile < MT_E; tile += gridDim.x * 8) {
    { // stage x[src] tile (each lane copies half a row) + dst indices
      int e = tile * 16 + n16;
      int s = esrc[e];
      if (lh == 0) dsts[wave][n16] = edst[e];
      const float4* sp = (const float4*)(x + (size_t)s * 32 + lh * 16);
      float4* dp = (float4*)&xs[wave][n16 * 32 + lh * 16];
      dp[0] = sp[0]; dp[1] = sp[1]; dp[2] = sp[2]; dp[3] = sp[3];
    }
    asm volatile("s_wait_dscnt 0x0" ::: "memory"); // same-wave LDS staging complete

    float acc0[8], acc1[8];
    #pragma unroll
    for (int r = 0; r < 8; ++r) { acc0[r] = 0.f; acc1[r] = 0.f; }

    for (int i = 0; i < 32; ++i) {
      float xsv[8];
      #pragma unroll
      for (int r = 0; r < 8; ++r) xsv[r] = xs[wave][(r + 8 * lh) * 32 + i];
      #pragma unroll
      for (int ohi = 0; ohi < 2; ++ohi) {
        v4u w = __builtin_nontemporal_load(
            (const v4u*)wpack + ((size_t)tile * 64 + 2 * i + ohi) * 32 + lane);
        float* acc = ohi ? acc1 : acc0;
        #pragma unroll
        for (int j = 0; j < 4; ++j) {
          unsigned dw = w[j];
          acc[2 * j]     += xsv[2 * j]     * bf2f((unsigned short)(dw & 0xFFFFu));
          acc[2 * j + 1] += xsv[2 * j + 1] * bf2f((unsigned short)(dw >> 16));
        }
      }
    }
    #pragma unroll
    for (int r = 0; r < 8; ++r) {
      int d = dsts[wave][r + 8 * lh];
      atomicAdd(&msum[(size_t)d * 32 + n16],      acc0[r]);
      atomicAdd(&msum[(size_t)d * 32 + 16 + n16], acc1[r]);
    }
  }
}

// m = relu(msum / max(cnt,1) + conv_bias)
__global__ __launch_bounds__(256) void k_mfin(const float* __restrict__ msum,
                                              const float* __restrict__ cnt,
                                              const float* __restrict__ cb,
                                              float* __restrict__ m) {
  size_t total = (size_t)NN * 32;
  for (size_t g = (size_t)blockIdx.x * 256 + threadIdx.x; g < total; g += (size_t)gridDim.x * 256) {
    size_t n = g >> 5; int o = (int)(g & 31);
    float v = msum[g] / fmaxf(cnt[n], 1.f) + cb[o];
    m[g] = fmaxf(v, 0.f);
  }
}

// ---------------------------------------------------------------------------
// GRU cell, WMMA: gi = m@Wih^T+bih, gh = h@Whh^T+bhh (K=32, one WMMA each tile)
__global__ __launch_bounds__(256) void k_gru(const float* __restrict__ m,
                                             const float* __restrict__ hprev,
                                             const float* __restrict__ wih,
                                             const float* __restrict__ whh,
                                             const float* __restrict__ bih,
                                             const float* __restrict__ bhh,
                                             float* __restrict__ hnext) {
  __shared__ unsigned short wihs[96 * 40];
  __shared__ unsigned short whhs[96 * 40];
  int tid = threadIdx.x;
  for (int idx = tid; idx < 96 * 32; idx += 256) {
    int rrow = idx >> 5, k = idx & 31;
    wihs[rrow * 40 + k] = f2bf(wih[idx]);
    whhs[rrow * 40 + k] = f2bf(whh[idx]);
  }
  __syncthreads();
  int lane = tid & 31, wave = tid >> 5, n16 = lane & 15, lh = lane >> 4;
  float bi[6], bh[6];
  #pragma unroll
  for (int nt = 0; nt < 6; ++nt) { bi[nt] = bih[nt * 16 + n16]; bh[nt] = bhh[nt * 16 + n16]; }

  for (int tile = blockIdx.x * 8 + wave; tile < MT_N; tile += gridDim.x * 8) {
    v16bf am = loadA_f32(m     + (size_t)(tile * 16 + n16) * 32, lh);
    v16bf ah = loadA_f32(hprev + (size_t)(tile * 16 + n16) * 32, lh);
    v8f gi[6], gh[6];
    #pragma unroll
    for (int nt = 0; nt < 6; ++nt) {
      v8f ci, ch;
      #pragma unroll
      for (int r = 0; r < 8; ++r) { ci[r] = bi[nt]; ch[r] = bh[nt]; }
      const unsigned short* bri = wihs + (nt * 16 + n16) * 40 + lh * 16;
      const unsigned short* brh = whhs + (nt * 16 + n16) * 40 + lh * 16;
      ABu ui, uh;
      ui.h[0] = *(const v8bf*)(bri); ui.h[1] = *(const v8bf*)(bri + 8);
      uh.h[0] = *(const v8bf*)(brh); uh.h[1] = *(const v8bf*)(brh + 8);
      gi[nt] = __builtin_amdgcn_wmma_f32_16x16x32_bf16(false, am, false, ui.v,
                                                       (short)0, ci, false, false);
      gh[nt] = __builtin_amdgcn_wmma_f32_16x16x32_bf16(false, ah, false, uh.v,
                                                       (short)0, ch, false, false);
    }
    #pragma unroll
    for (int p2 = 0; p2 < 2; ++p2) {
      #pragma unroll
      for (int r = 0; r < 8; ++r) {
        int row = tile * 16 + r + 8 * lh;
        int o   = p2 * 16 + n16;
        float rg = sigm(gi[0 + p2][r] + gh[0 + p2][r]);
        float zg = sigm(gi[2 + p2][r] + gh[2 + p2][r]);
        float ng = tanhf(gi[4 + p2][r] + rg * gh[4 + p2][r]);
        float hv = hprev[(size_t)row * 32 + o];
        hnext[(size_t)row * 32 + o] = (1.f - zg) * ng + zg * hv;
      }
    }
  }
}

// ---------------------------------------------------------------------------
// Set2Set pieces
__global__ __launch_bounds__(256) void k_lstm(const float* __restrict__ qstar,
                                              float* __restrict__ hs, float* __restrict__ cs,
                                              const float* __restrict__ wih,   // [128,64]
                                              const float* __restrict__ whh,   // [128,32]
                                              const float* __restrict__ bih,
                                              const float* __restrict__ bhh) {
  __shared__ float qs[2][64], hh[2][32], g[2][128];
  int tid = threadIdx.x, gl = tid >> 7, gt = tid & 127;
  int b = blockIdx.x * 2 + gl;
  if (b < NB) {
    if (gt < 64) qs[gl][gt] = qstar[(size_t)b * 64 + gt];
    else if (gt < 96) hh[gl][gt - 64] = hs[(size_t)b * 32 + (gt - 64)];
  }
  __syncthreads();
  if (b < NB) {
    float acc = bih[gt] + bhh[gt];
    #pragma unroll 16
    for (int k = 0; k < 64; ++k) acc += qs[gl][k] * wih[gt * 64 + k];
    #pragma unroll 16
    for (int k = 0; k < 32; ++k) acc += hh[gl][k] * whh[gt * 32 + k];
    g[gl][gt] = acc;
  }
  __syncthreads();
  if (b < NB && gt < 32) {
    float ig = sigm(g[gl][gt]);
    float fg = sigm(g[gl][32 + gt]);
    float gg = tanhf(g[gl][64 + gt]);
    float og = sigm(g[gl][96 + gt]);
    float c  = fg * cs[(size_t)b * 32 + gt] + ig * gg;
    cs[(size_t)b * 32 + gt] = c;
    hs[(size_t)b * 32 + gt] = og * tanhf(c);   // h = q
  }
}

__global__ __launch_bounds__(256) void k_att_e(const float* __restrict__ x,
                                               const float* __restrict__ q,
                                               const int* __restrict__ gidx,
                                               float* __restrict__ evec,
                                               unsigned* __restrict__ ekeys) {
  for (int n = blockIdx.x * 256 + threadIdx.x; n < NN; n += gridDim.x * 256) {
    int b = gidx[n];
    const float* xr = x + (size_t)n * 32;
    const float* qr = q + (size_t)b * 32;
    float e = 0.f;
    #pragma unroll
    for (int k = 0; k < 32; ++k) e += xr[k] * qr[k];
    evec[n] = e;
    atomicMax(&ekeys[b], fkey(e));
  }
}

__global__ __launch_bounds__(256) void k_att_exp(const float* __restrict__ evec,
                                                 const unsigned* __restrict__ ekeys,
                                                 const int* __restrict__ gidx,
                                                 float* __restrict__ exvec,
                                                 float* __restrict__ denom) {
  for (int n = blockIdx.x * 256 + threadIdx.x; n < NN; n += gridDim.x * 256) {
    int b = gidx[n];
    float ex = __expf(evec[n] - fdec(ekeys[b]));
    exvec[n] = ex;
    atomicAdd(&denom[b], ex);
  }
}

__global__ __launch_bounds__(256) void k_att_r(const float* __restrict__ x,
                                               const float* __restrict__ exvec,
                                               const float* __restrict__ denom,
                                               const int* __restrict__ gidx,
                                               float* __restrict__ rbuf) {
  size_t total = (size_t)NN * 32;
  for (size_t g = (size_t)blockIdx.x * 256 + threadIdx.x; g < total; g += (size_t)gridDim.x * 256) {
    size_t n = g >> 5; int o = (int)(g & 31);
    int b = gidx[n];
    float a = exvec[n] / denom[b];
    atomicAdd(&rbuf[(size_t)b * 32 + o], a * x[g]);
  }
}

__global__ __launch_bounds__(256) void k_qstar(const float* __restrict__ hs,
                                               const float* __restrict__ rbuf,
                                               float* __restrict__ qstar) {
  int total = NB * 64;
  for (int g = blockIdx.x * 256 + threadIdx.x; g < total; g += gridDim.x * 256) {
    int b = g >> 6, j = g & 63;
    qstar[g] = (j < 32) ? hs[b * 32 + j] : rbuf[b * 32 + (j - 32)];
  }
}

__global__ __launch_bounds__(256) void k_out(const float* __restrict__ qstar,
                                             const float* __restrict__ hfin,
                                             float* __restrict__ out) {
  size_t total = (size_t)NB * 64 + (size_t)NN * 32;
  for (size_t g = (size_t)blockIdx.x * 256 + threadIdx.x; g < total; g += (size_t)gridDim.x * 256) {
    out[g] = (g < (size_t)NB * 64) ? qstar[g] : hfin[g - (size_t)NB * 64];
  }
}

// ---------------------------------------------------------------------------
static inline unsigned gz(size_t n) {
  size_t b = (n + 255) / 256;
  return (unsigned)(b < 4096 ? (b ? b : 1) : 4096);
}

extern "C" void kernel_launch(void* const* d_in, const int* in_sizes, int n_in,
                              void* d_out, int out_size, void* d_ws, size_t ws_size,
                              hipStream_t stream) {
  (void)in_sizes; (void)n_in; (void)out_size; (void)ws_size;
  const float* nf      = (const float*)d_in[0];
  const float* ef      = (const float*)d_in[1];
  const float* lin0_w  = (const float*)d_in[2];
  const float* lin0_b  = (const float*)d_in[3];
  const float* nn_w1   = (const float*)d_in[4];
  const float* nn_b1   = (const float*)d_in[5];
  const float* nn_w2   = (const float*)d_in[6];
  const float* nn_b2   = (const float*)d_in[7];
  const float* convb   = (const float*)d_in[8];
  const float* gru_wih = (const float*)d_in[9];
  const float* gru_whh = (const float*)d_in[10];
  const float* gru_bih = (const float*)d_in[11];
  const float* gru_bhh = (const float*)d_in[12];
  const float* l_wih   = (const float*)d_in[13];
  const float* l_whh   = (const float*)d_in[14];
  const float* l_bih   = (const float*)d_in[15];
  const float* l_bhh   = (const float*)d_in[16];
  const int*   eidx    = (const int*)d_in[17];
  const int*   gidx    = (const int*)d_in[18];
  const int* esrc = eidx;
  const int* edst = eidx + NE;

  // workspace carve (requires ~650 MB)
  char* p = (char*)d_ws;
  auto take = [&](size_t bytes) -> char* {
    char* r = p; p += (bytes + 255) & ~(size_t)255; return r;
  };
  unsigned short* wpack = (unsigned short*)take((size_t)NE * 1024 * 2);   // 512 MB bf16
  unsigned short* hwb   = (unsigned short*)take((size_t)NE * HID * 2);    //  64 MB bf16
  float* x0    = (float*)take((size_t)NN * 32 * 4);
  float* hA    = (float*)take((size_t)NN * 32 * 4);
  float* hB    = (float*)take((size_t)NN * 32 * 4);
  float* msum  = (float*)take((size_t)NN * 32 * 4);
  float* mbuf  = (float*)take((size_t)NN * 32 * 4);
  float* cnt   = (float*)take((size_t)NN * 4);
  float* evec  = (float*)take((size_t)NN * 4);
  float* exvec = (float*)take((size_t)NN * 4);
  unsigned* ekeys = (unsigned*)take((size_t)NB * 4);
  float* denom = (float*)take((size_t)NB * 4);
  float* hs    = (float*)take((size_t)NB * 32 * 4);
  float* cs    = (float*)take((size_t)NB * 32 * 4);
  float* rbuf  = (float*)take((size_t)NB * 32 * 4);
  float* qstar = (float*)take((size_t)NB * 64 * 4);

  // ---- precompute (edge network is loop-invariant) ----
  k_zero<<<gz(NN), 256, 0, stream>>>((unsigned*)cnt, NN);
  k_lin0<<<1024, 256, 0, stream>>>(nf, lin0_w, lin0_b, x0);
  k_hw<<<4096, 256, 0, stream>>>(ef, nn_w1, nn_b1, hwb);
  {
    dim3 g(512, 8);
    k_edge_gemm<<<g, 256, 0, stream>>>(hwb, nn_w2, nn_b2, wpack);
  }
  k_degree<<<1024, 256, 0, stream>>>(edst, cnt);

  // ---- 3 message-passing iterations ----
  const float* cur = x0;
  float* pong[2] = {hA, hB};
  for (int it = 0; it < 3; ++it) {
    k_zero<<<gz((size_t)NN * 32), 256, 0, stream>>>((unsigned*)msum, (size_t)NN * 32);
    k_conv<<<1024, 256, 0, stream>>>(wpack, cur, esrc, edst, msum);
    k_mfin<<<gz((size_t)NN * 32), 256, 0, stream>>>(msum, cnt, convb, mbuf);
    float* hn = pong[it & 1];
    k_gru<<<512, 256, 0, stream>>>(mbuf, cur, gru_wih, gru_whh, gru_bih, gru_bhh, hn);
    cur = hn;
  }

  // ---- Set2Set (3 steps) ----
  k_zero<<<gz((size_t)NB * 32), 256, 0, stream>>>((unsigned*)hs, (size_t)NB * 32);
  k_zero<<<gz((size_t)NB * 32), 256, 0, stream>>>((unsigned*)cs, (size_t)NB * 32);
  k_zero<<<gz((size_t)NB * 64), 256, 0, stream>>>((unsigned*)qstar, (size_t)NB * 64);
  for (int s = 0; s < 3; ++s) {
    k_lstm<<<(NB + 1) / 2, 256, 0, stream>>>(qstar, hs, cs, l_wih, l_whh, l_bih, l_bhh);
    k_zero<<<gz(NB), 256, 0, stream>>>(ekeys, NB);
    k_zero<<<gz(NB), 256, 0, stream>>>((unsigned*)denom, NB);
    k_zero<<<gz((size_t)NB * 32), 256, 0, stream>>>((unsigned*)rbuf, (size_t)NB * 32);
    k_att_e<<<gz(NN), 256, 0, stream>>>(cur, hs, gidx, evec, ekeys);
    k_att_exp<<<gz(NN), 256, 0, stream>>>(evec, ekeys, gidx, exvec, denom);
    k_att_r<<<gz((size_t)NN * 32), 256, 0, stream>>>(cur, exvec, denom, gidx, rbuf);
    k_qstar<<<gz((size_t)NB * 64), 256, 0, stream>>>(hs, rbuf, qstar);
  }

  k_out<<<gz((size_t)NB * 64 + (size_t)NN * 32), 256, 0, stream>>>(qstar, cur, (float*)d_out);
}
